// TemporalGATTransformer_67388036874459
// MI455X (gfx1250) — matrix operand order repaired and checked by reading, not compiled, not run on hardware
//
#include <hip/hip_runtime.h>
#include <math.h>

#define HID   128
#define HEADS 8
#define TT    5
#define FIN   32

// GEMM tiling: block tile 64(M) x 128(N), K-step 32, 8 waves of 16x64 each
#define BM 64
#define BN 128
#define BK 32
#define APAD 4
#define BPAD 4

typedef __attribute__((ext_vector_type(16))) _Float16 v16h;
typedef __attribute__((ext_vector_type(8)))  float    v8f;

#if defined(__gfx1250__) && __has_builtin(__builtin_amdgcn_tensor_load_to_lds) && \
    __has_builtin(__builtin_amdgcn_s_wait_tensorcnt)
#define GEMM_USE_TDM 1
typedef __attribute__((ext_vector_type(4))) unsigned int u32x4;
typedef __attribute__((ext_vector_type(8))) int          i32x8;
typedef __attribute__((ext_vector_type(4))) int          i32x4;

// Issue one 2D-tile TDM load (f32 elements) into LDS with row padding so the
// LDS layout matches our bank-padded shared arrays (pad_amount = 4 DWORDs).
__device__ __forceinline__ void tdm_load_2d(const void* gaddr_, unsigned lds,
                                            unsigned rowlen,   // tile/tensor dim0 (f32 elems)
                                            unsigned nrows,    // tile dim1
                                            unsigned validRows,// tensor dim1 (OOB rows -> 0)
                                            unsigned long long rowStride,
                                            unsigned padIntervalCode)
{
    unsigned long long ga = (unsigned long long)gaddr_;
    u32x4 g0;
    g0[0] = 1u;                                            // count=1
    g0[1] = lds;                                           // lds_addr
    g0[2] = (unsigned)ga;                                  // global_addr lo
    g0[3] = (unsigned)((ga >> 32) & 0x1FFFFFFu) | (2u << 30); // type=2
    i32x8 g1;
    g1[0] = (int)((2u << 16)                // data_size = 4B
                | (1u << 20)                // pad_enable
                | (padIntervalCode << 22)   // DWORDs per row before pad
                | (3u << 25));              // pad_amount = 4 DWORDs
    g1[1] = (int)((rowlen & 0xFFFFu) << 16);                       // tensor_dim0 lo16
    g1[2] = (int)(((rowlen >> 16) & 0xFFFFu) | ((validRows & 0xFFFFu) << 16));
    g1[3] = (int)(((validRows >> 16) & 0xFFFFu) | ((rowlen & 0xFFFFu) << 16)); // tile_dim0
    g1[4] = (int)(nrows & 0xFFFFu);                                // tile_dim1 (dim2=0)
    g1[5] = (int)(unsigned)(rowStride & 0xFFFFFFFFu);              // dim0_stride lo
    g1[6] = (int)(unsigned)((rowStride >> 32) & 0xFFFFu);          // dim0_stride hi
    g1[7] = 0;
    i32x4 z4 = {0, 0, 0, 0};
#if __clang_major__ >= 23
    i32x8 z8 = {0, 0, 0, 0, 0, 0, 0, 0};
    __builtin_amdgcn_tensor_load_to_lds(g0, g1, z4, z4, z8, 0);
#else
    __builtin_amdgcn_tensor_load_to_lds(g0, g1, z4, z4, 0);
#endif
}
#else
#define GEMM_USE_TDM 0
#endif

// ---------------------------------------------------------------------------
// Double-buffered, LDS-staged WMMA GEMM:
//   D(MxN) = act( A(MxK)*B(KxN) + bias + residual )
// Per K-step the 64x32 A tile and 32x128 B tile are staged in LDS (f32).
// TDM path: wave 0 issues async tensor_load_to_lds for stage i+1 BEFORE the
// WMMAs of stage i, then s_wait_tensorcnt 0 + barrier (DMA/compute overlap).
// Fallback: cooperative float4 loads. Each wave computes 16x64 -> 4 WMMA per
// K-step sharing one A fragment. f32->f16 conversion at fragment build, f32
// accumulation. N % 128 == 0, K % 32 == 0; M ragged (zero-fill / clamp loads,
// guarded stores). No early exit: EXEC all-ones at every WMMA.
// ---------------------------------------------------------------------------
__global__ void __launch_bounds__(256)
wmma_gemm_f16(const float* __restrict__ A, int lda,
              const float* __restrict__ B, int ldb,
              float* __restrict__ D, int ldd,
              const float* __restrict__ bias,
              const float* __restrict__ residual,
              int M, int N, int K, int relu)
{
    __shared__ float As[2][BM][BK + APAD];   // row stride 36 f32 (16B aligned)
    __shared__ float Bs[2][BK][BN + BPAD];   // row stride 132 f32 (16B aligned)

    const int tid    = threadIdx.x;
    const int lane   = tid & 31;
    const int wv     = tid >> 5;             // 0..7
    const int tilesN = N / BN;
    const int m0     = (blockIdx.x / tilesN) * BM;
    const int n0     = (blockIdx.x % tilesN) * BN;

    const int wm  = wv >> 1;                 // 0..3 : 16-row subtile
    const int wn  = wv & 1;                  // 0..1 : 64-col subtile
    const int row = lane & 15;
    const int hi  = lane >> 4;

    auto stage_issue = [&](int st, int k0) {
#if GEMM_USE_TDM
        if (wv == 0) {
            tdm_load_2d(A + (size_t)m0 * lda + k0,
                        (unsigned)(unsigned long long)&As[st][0][0],
                        BK, BM, (unsigned)(M - m0), (unsigned long long)lda,
                        4u /* 32 DWORDs/row */);
            tdm_load_2d(B + (size_t)k0 * ldb + n0,
                        (unsigned)(unsigned long long)&Bs[st][0][0],
                        BN, BK, BK, (unsigned long long)ldb,
                        6u /* 128 DWORDs/row */);
        }
#else
        // A: 64x32 f32 = 512 float4, 2 per thread
#pragma unroll
        for (int r = 0; r < (BM * BK) / (256 * 4); ++r) {
            int e  = r * 256 + tid;
            int am = e >> 3, a4 = e & 7;
            int ga = min(m0 + am, M - 1);
            float4 v = *(const float4*)(A + (size_t)ga * lda + k0 + a4 * 4);
            *(float4*)&As[st][am][a4 * 4] = v;
        }
        // B: 32x128 f32 = 1024 float4, 4 per thread
#pragma unroll
        for (int r = 0; r < (BK * BN) / (256 * 4); ++r) {
            int e  = r * 256 + tid;
            int bk = e >> 5, b4 = e & 31;
            float4 v = *(const float4*)(B + (size_t)(k0 + bk) * ldb + n0 + b4 * 4);
            *(float4*)&Bs[st][bk][b4 * 4] = v;
        }
#endif
    };
    auto stage_wait = [&]() {
#if GEMM_USE_TDM
        if (wv == 0) __builtin_amdgcn_s_wait_tensorcnt(0);
#endif
    };

    v8f acc[4] = {};
    const int ar = wm * 16 + row;
    const int nk = K / BK;

    stage_issue(0, 0);
    stage_wait();
    __syncthreads();

    for (int i = 0; i < nk; ++i) {
        const int cur = i & 1;
        if (i + 1 < nk) stage_issue(cur ^ 1, (i + 1) * BK);

        // A fragment: 16x32, contiguous k = hi*16 + s (same order as B), f32->f16 here
        v16h a;
#pragma unroll
        for (int p = 0; p < 4; ++p) {
            const int ka = hi * 16 + 2 * p;
            a[2 * p]     = (_Float16)As[cur][ar][ka];
            a[2 * p + 1] = (_Float16)As[cur][ar][ka + 1];
            a[2 * p + 8] = (_Float16)As[cur][ar][ka + 8];
            a[2 * p + 9] = (_Float16)As[cur][ar][ka + 9];
        }
        // 4 B fragments (16-bit 32x16 layout), one WMMA each, A reused
#pragma unroll
        for (int t = 0; t < 4; ++t) {
            const int col = wn * 64 + t * 16 + row;
            v16h b;
#pragma unroll
            for (int p = 0; p < 8; ++p) {
                const int kb = hi * 16 + 2 * p;
                b[2 * p]     = (_Float16)Bs[cur][kb][col];
                b[2 * p + 1] = (_Float16)Bs[cur][kb + 1][col];
            }
            acc[t] = __builtin_amdgcn_wmma_f32_16x16x32_f16(false, a, false, b,
                                                            (short)0, acc[t],
                                                            false, false);
        }
        stage_wait();
        __syncthreads();
    }

    // epilogue: C/D layout VGPR i -> row i + hi*8, col lane&15
#pragma unroll
    for (int t = 0; t < 4; ++t) {
        const int dcol = n0 + wn * 64 + t * 16 + row;
#pragma unroll
        for (int i = 0; i < 8; ++i) {
            const int drow = m0 + wm * 16 + i + hi * 8;
            if (drow < M) {
                float v = acc[t][i];
                if (bias)     v += bias[dcol];
                if (residual) v += residual[(size_t)drow * ldd + dcol];
                if (relu)     v = v > 0.f ? v : 0.f;
                D[(size_t)drow * ldd + dcol] = v;
            }
        }
    }
}

// ---------------------------------------------------------------------------
// Scatter / softmax / normalization helper kernels (latency-bound VALU work)
// ---------------------------------------------------------------------------
__global__ void fill_k(float* __restrict__ p, float v, long n) {
    long i = (long)blockIdx.x * blockDim.x + threadIdx.x;
    if (i < n) p[i] = v;
}

__device__ __forceinline__ void atomicMaxF(float* addr, float val) {
    if (val >= 0.f) atomicMax((int*)addr, __float_as_int(val));
    else            atomicMin((unsigned int*)addr, __float_as_uint(val));
}

__global__ void gat_att_k(const float* __restrict__ z,
                          const float* __restrict__ att_src,
                          const float* __restrict__ att_dst,
                          float* __restrict__ a_s, float* __restrict__ a_d, int Nn)
{
    int idx = blockIdx.x * blockDim.x + threadIdx.x;
    if (idx >= Nn * HEADS) return;
    int n = idx / HEADS, h = idx % HEADS;
    const float* zp = z + (size_t)n * (HEADS * HID) + h * HID;
    const float* as = att_src + h * HID;
    const float* ad = att_dst + h * HID;
    float ss = 0.f, sd = 0.f;
    for (int c = 0; c < HID; ++c) { float zv = zp[c]; ss += zv * as[c]; sd += zv * ad[c]; }
    a_s[idx] = ss; a_d[idx] = sd;
}

__device__ __forceinline__ void edge_sd(const int* ei, int E, int e, int& src, int& dst) {
    if (e < E) { src = ei[e]; dst = ei[E + e]; }
    else       { src = e - E; dst = e - E; }        // self loops
}

__global__ void edge_max_k(const int* __restrict__ ei, int E, int ET,
                           const float* __restrict__ a_s, const float* __restrict__ a_d,
                           float* __restrict__ m)
{
    int idx = blockIdx.x * blockDim.x + threadIdx.x;
    if (idx >= ET * HEADS) return;
    int e = idx / HEADS, h = idx % HEADS, src, dst;
    edge_sd(ei, E, e, src, dst);
    float x = a_s[src * HEADS + h] + a_d[dst * HEADS + h];
    float lv = x >= 0.f ? x : 0.2f * x;             // leaky_relu(0.2)
    atomicMaxF(&m[dst * HEADS + h], lv);
}

__global__ void edge_denom_k(const int* __restrict__ ei, int E, int ET,
                             const float* __restrict__ a_s, const float* __restrict__ a_d,
                             const float* __restrict__ m, float* __restrict__ denom)
{
    int idx = blockIdx.x * blockDim.x + threadIdx.x;
    if (idx >= ET * HEADS) return;
    int e = idx / HEADS, h = idx % HEADS, src, dst;
    edge_sd(ei, E, e, src, dst);
    float x = a_s[src * HEADS + h] + a_d[dst * HEADS + h];
    float lv = x >= 0.f ? x : 0.2f * x;
    atomicAdd(&denom[dst * HEADS + h], __expf(lv - m[dst * HEADS + h]));
}

__global__ void edge_acc_k(const int* __restrict__ ei, int E, int ET,
                           const float* __restrict__ a_s, const float* __restrict__ a_d,
                           const float* __restrict__ m, const float* __restrict__ denom,
                           const float* __restrict__ z, float* __restrict__ hout)
{
    long idx = (long)blockIdx.x * blockDim.x + threadIdx.x;
    if (idx >= (long)ET * HID) return;
    int e = (int)(idx >> 7), c = (int)(idx & (HID - 1)), src, dst;
    edge_sd(ei, E, e, src, dst);
    float acc = 0.f;
#pragma unroll
    for (int h = 0; h < HEADS; ++h) {
        float x  = a_s[src * HEADS + h] + a_d[dst * HEADS + h];
        float lv = x >= 0.f ? x : 0.2f * x;
        float w  = __expf(lv - m[dst * HEADS + h]);
        float alpha = w / (denom[dst * HEADS + h] + 1e-16f);
        acc += alpha * z[(size_t)src * (HEADS * HID) + h * HID + c];
    }
    atomicAdd(&hout[(size_t)dst * HID + c], acc * (1.f / HEADS));
}

__global__ void bias_relu_k(float* __restrict__ hp, const float* __restrict__ bias, int Nn) {
    long idx = (long)blockIdx.x * blockDim.x + threadIdx.x;
    if (idx >= (long)Nn * HID) return;
    float v = hp[idx] + bias[idx & (HID - 1)];
    hp[idx] = v > 0.f ? v : 0.f;
}

__global__ void pool_sum_k(const float* __restrict__ h, const int* __restrict__ assign,
                           float* __restrict__ sums, float* __restrict__ cnt, int Nn) {
    long idx = (long)blockIdx.x * blockDim.x + threadIdx.x;
    if (idx >= (long)Nn * HID) return;
    int n = (int)(idx >> 7), c = (int)(idx & (HID - 1));
    int g = assign[n];
    atomicAdd(&sums[(size_t)g * HID + c], h[idx]);
    if (c == 0) atomicAdd(&cnt[g], 1.f);
}

__global__ void pool_div_k(float* __restrict__ spat, const float* __restrict__ sums,
                           const float* __restrict__ cnt, int NIc) {
    int idx = blockIdx.x * blockDim.x + threadIdx.x;
    if (idx >= NIc * HID) return;
    spat[idx] = sums[idx] / fmaxf(cnt[idx >> 7], 1.f);
}

__global__ void build_enc_k(const float* __restrict__ history,
                            const float* __restrict__ spat,
                            float* __restrict__ zseq, int NIc)
{
    int idx = blockIdx.x * blockDim.x + threadIdx.x;
    if (idx >= NIc * TT * HID) return;
    int d = idx & (HID - 1);
    int s = (idx >> 7) % TT;
    int b = idx / (TT * HID);
    float v = (s < TT - 1) ? history[((size_t)(s + 1) * NIc + b) * HID + d]
                           : spat[(size_t)b * HID + d];
    int   i   = d >> 1;
    float dv  = __expf(-(float)(2 * i) * (logf(10000.f) / (float)HID));
    float ang = (float)s * dv;
    float pe  = (d & 1) ? __cosf(ang) : __sinf(ang);
    zseq[idx] = v + pe;
}

__global__ void attn_k(const float* __restrict__ qkv, float* __restrict__ o, int NIc) {
    int b = blockIdx.x, t = threadIdx.x;
    if (t >= HEADS * TT) return;
    int h = t / TT, qs = t % TT;
    const float* qp = qkv + (size_t)(b * TT + qs) * 384 + h * 16;
    float sc[TT]; float mx = -1e30f;
    for (int ks = 0; ks < TT; ++ks) {
        const float* kp = qkv + (size_t)(b * TT + ks) * 384 + 128 + h * 16;
        float sdot = 0.f;
        for (int d = 0; d < 16; ++d) sdot += qp[d] * kp[d];
        sc[ks] = sdot * 0.25f;
        mx = fmaxf(mx, sc[ks]);
    }
    float den = 0.f;
    for (int ks = 0; ks < TT; ++ks) { sc[ks] = __expf(sc[ks] - mx); den += sc[ks]; }
    float inv = 1.f / den;
    for (int d = 0; d < 16; ++d) {
        float acc = 0.f;
        for (int ks = 0; ks < TT; ++ks)
            acc += sc[ks] * qkv[(size_t)(b * TT + ks) * 384 + 256 + h * 16 + d];
        o[(size_t)(b * TT + qs) * HID + h * 16 + d] = acc * inv;
    }
}

__global__ void layer_norm_k(const float* __restrict__ x, const float* __restrict__ s,
                             const float* __restrict__ bsh, float* __restrict__ out) {
    __shared__ float red[HID];
    int r = blockIdx.x, t = threadIdx.x;
    float v = x[(size_t)r * HID + t];
    red[t] = v; __syncthreads();
    for (int o = HID / 2; o > 0; o >>= 1) { if (t < o) red[t] += red[t + o]; __syncthreads(); }
    float mu = red[0] * (1.f / HID); __syncthreads();
    float d = v - mu;
    red[t] = d * d; __syncthreads();
    for (int o = HID / 2; o > 0; o >>= 1) { if (t < o) red[t] += red[t + o]; __syncthreads(); }
    float var = red[0] * (1.f / HID);
    out[(size_t)r * HID + t] = d * rsqrtf(var + 1e-5f) * s[t] + bsh[t];
}

__global__ void build_comb_k(const float* __restrict__ spat, const float* __restrict__ zseq,
                             float* __restrict__ comb, int NIc) {
    int idx = blockIdx.x * blockDim.x + threadIdx.x;
    if (idx >= NIc * 2 * HID) return;
    int b = idx >> 8, d = idx & 255;
    comb[idx] = (d < HID) ? spat[(size_t)b * HID + d]
                          : zseq[(size_t)(b * TT + (TT - 1)) * HID + (d - HID)];
}

__global__ void mlp_head_k(const float* __restrict__ hm, const float* __restrict__ W2,
                           const float* __restrict__ b2, float* __restrict__ out) {
    __shared__ float red[HID];
    int r = blockIdx.x, t = threadIdx.x;
    red[t] = hm[(size_t)r * HID + t] * W2[t]; __syncthreads();
    for (int o = HID / 2; o > 0; o >>= 1) { if (t < o) red[t] += red[t + o]; __syncthreads(); }
    if (t == 0) out[r] = red[0] + b2[0];
}

// ---------------------------------------------------------------------------
// Host side
// ---------------------------------------------------------------------------
static inline void gemm(const float* A, int lda, const float* B, int ldb,
                        float* D, int ldd, int M, int N, int K,
                        const float* bias, const float* residual, int relu,
                        hipStream_t stream)
{
    int blocks = ((M + BM - 1) / BM) * (N / BN);
    wmma_gemm_f16<<<blocks, 256, 0, stream>>>(A, lda, B, ldb, D, ldd,
                                              bias, residual, M, N, K, relu);
}
static inline int g1(long n) { return (int)((n + 255) / 256); }

extern "C" void kernel_launch(void* const* d_in, const int* in_sizes, int n_in,
                              void* d_out, int out_size, void* d_ws, size_t ws_size,
                              hipStream_t stream)
{
    const float* x       = (const float*)d_in[0];
    const int*   ei      = (const int*)  d_in[1];
    const int*   assign  = (const int*)  d_in[2];
    const float* history = (const float*)d_in[3];
    const float* prm[40];
    for (int i = 0; i < n_in - 4 && i < 40; ++i) prm[i] = (const float*)d_in[4 + i];
    // params flattened in insertion order:
    // 0..3 gat0{W,att_src,att_dst,bias} 4..7 gat1{...}
    // 8..19 tr0{Wqkv,bqkv,Wo,bo,ln1_s,ln1_b,W1,b1,W2,b2,ln2_s,ln2_b} 20..31 tr1
    // 32..35 actor{W1,b1,W2,b2} 36..39 critic{...}
    const int N  = in_sizes[0] / FIN;
    const int E  = in_sizes[1] / 2;
    const int NI = in_sizes[3] / (TT * HID);
    const int ET = E + N;
    const int Mt = NI * TT;

    float* w = (float*)d_ws;
    float* Z    = w; w += (size_t)N * HEADS * HID;
    float* H1   = w; w += (size_t)N * HID;
    float* H2   = w; w += (size_t)N * HID;
    float* AS   = w; w += (size_t)N * HEADS;
    float* AD   = w; w += (size_t)N * HEADS;
    float* Mx   = w; w += (size_t)N * HEADS;
    float* Dn   = w; w += (size_t)N * HEADS;
    float* sums = w; w += (size_t)NI * HID;
    float* cnt  = w; w += (size_t)NI;
    float* spat = w; w += (size_t)NI * HID;
    float* zseq = w; w += (size_t)Mt * HID;
    float* qkv  = w; w += (size_t)Mt * 3 * HID;
    float* attO = w; w += (size_t)Mt * HID;
    float* tmp  = w; w += (size_t)Mt * HID;
    float* xn   = w; w += (size_t)Mt * HID;
    float* y1   = w; w += (size_t)Mt * 4 * HID;
    float* comb = w; w += (size_t)NI * 2 * HID;
    float* hA   = w; w += (size_t)NI * HID;
    float* hC   = w; w += (size_t)NI * HID;

    // ---- GAT layers -------------------------------------------------------
    const float* hin = x; int fin = FIN;
    float* houts[2] = {H1, H2};
    for (int L = 0; L < 2; ++L) {
        const float* W   = prm[L * 4 + 0];
        const float* asv = prm[L * 4 + 1];
        const float* adv = prm[L * 4 + 2];
        const float* bv  = prm[L * 4 + 3];
        float* hout = houts[L];
        gemm(hin, fin, W, HEADS * HID, Z, HEADS * HID, N, HEADS * HID, fin,
             nullptr, nullptr, 0, stream);
        gat_att_k<<<g1((long)N * HEADS), 256, 0, stream>>>(Z, asv, adv, AS, AD, N);
        fill_k<<<g1((long)N * HEADS), 256, 0, stream>>>(Mx, -INFINITY, (long)N * HEADS);
        fill_k<<<g1((long)N * HEADS), 256, 0, stream>>>(Dn, 0.f, (long)N * HEADS);
        fill_k<<<g1((long)N * HID), 256, 0, stream>>>(hout, 0.f, (long)N * HID);
        edge_max_k<<<g1((long)ET * HEADS), 256, 0, stream>>>(ei, E, ET, AS, AD, Mx);
        edge_denom_k<<<g1((long)ET * HEADS), 256, 0, stream>>>(ei, E, ET, AS, AD, Mx, Dn);
        edge_acc_k<<<g1((long)ET * HID), 256, 0, stream>>>(ei, E, ET, AS, AD, Mx, Dn, Z, hout);
        bias_relu_k<<<g1((long)N * HID), 256, 0, stream>>>(hout, bv, N);
        hin = hout; fin = HID;
    }

    // ---- interval pooling -------------------------------------------------
    fill_k<<<g1((long)NI * HID), 256, 0, stream>>>(sums, 0.f, (long)NI * HID);
    fill_k<<<g1((long)NI), 256, 0, stream>>>(cnt, 0.f, (long)NI);
    pool_sum_k<<<g1((long)N * HID), 256, 0, stream>>>(H2, assign, sums, cnt, N);
    pool_div_k<<<g1((long)NI * HID), 256, 0, stream>>>(spat, sums, cnt, NI);

    // ---- temporal encoder -------------------------------------------------
    build_enc_k<<<g1((long)Mt * HID), 256, 0, stream>>>(history, spat, zseq, NI);
    for (int p = 0; p < 2; ++p) {
        const float** tp = &prm[8 + p * 12];
        gemm(zseq, HID, tp[0], 3 * HID, qkv, 3 * HID, Mt, 3 * HID, HID,
             tp[1], nullptr, 0, stream);
        attn_k<<<NI, 64, 0, stream>>>(qkv, attO, NI);
        gemm(attO, HID, tp[2], HID, tmp, HID, Mt, HID, HID, tp[3], zseq, 0, stream);
        layer_norm_k<<<Mt, HID, 0, stream>>>(tmp, tp[4], tp[5], xn);
        gemm(xn, HID, tp[6], 4 * HID, y1, 4 * HID, Mt, 4 * HID, HID, tp[7], nullptr, 1, stream);
        gemm(y1, 4 * HID, tp[8], HID, tmp, HID, Mt, HID, 4 * HID, tp[9], xn, 0, stream);
        layer_norm_k<<<Mt, HID, 0, stream>>>(tmp, tp[10], tp[11], zseq);
    }

    // ---- heads ------------------------------------------------------------
    build_comb_k<<<g1((long)NI * 2 * HID), 256, 0, stream>>>(spat, zseq, comb, NI);
    float* logits = (float*)d_out;
    float* value  = (float*)d_out + NI;
    gemm(comb, 2 * HID, prm[32], HID, hA, HID, NI, HID, 2 * HID, prm[33], nullptr, 1, stream);
    mlp_head_k<<<NI, HID, 0, stream>>>(hA, prm[34], prm[35], logits);
    gemm(comb, 2 * HID, prm[36], HID, hC, HID, NI, HID, 2 * HID, prm[37], nullptr, 1, stream);
    mlp_head_k<<<NI, HID, 0, stream>>>(hC, prm[38], prm[39], value);
    (void)ws_size; (void)out_size;
}
